// SelfAttention_79001628442778
// MI455X (gfx1250) — compile-verified, matrix operand order checked by reading, not run on hardware
//
#include <hip/hip_runtime.h>

typedef __attribute__((ext_vector_type(16))) _Float16 v16h;
typedef __attribute__((ext_vector_type(8)))  _Float16 v8h;
typedef __attribute__((ext_vector_type(4)))  _Float16 v4h;
typedef __attribute__((ext_vector_type(8)))  float    v8f;
typedef __attribute__((ext_vector_type(4)))  float    v4f;

#define N_TOK 64
#define DIM   256
#define DH    32
#define NH    8

// LDS half-index offsets. Strides padded so every 16-byte fragment load is
// 16B-aligned and lane->bank mapping is conflict-free (stride 264 halves =
// 132 dwords == 4 mod 64; stride 72 halves = 36 dwords == 36 mod 64).
#define SQ_OFF   0         // 64 rows x stride 264 = 16896 halves (phase 0/1)
#define SK_OFF   16896     // 64 rows x stride 264 = 16896 halves (phase 0/1)
#define SP_OFF   0         // 8 heads x 64 x 72    = 36864 halves (overlays Q/K after barrier)
#define SVT_OFF  36864     // 256 cols x stride 72 = 18432 halves (V transposed)
#define SY_OFF   55296     // 64 rows x stride 264 = 16896 halves
#define SMEM_HALVES 72192  // 144384 bytes total

__device__ __forceinline__ v8f wmma16(v16h a, v16h b, v8f c) {
  return __builtin_amdgcn_wmma_f32_16x16x32_f16(false, a, false, b, (short)0, c,
                                                false, false);
}

// Fragment loader for row-major [row][k] LDS data (A fragments; also B
// fragments when stored K-contiguous with "row" = output column).
// ISA 16-bit layout: lanes 0-15 -> row row_base+lane, halves 0..7 = k kbase..+7,
// halves 8..15 = k kbase+16..+23; lanes 16-31 -> same rows, k ranges shifted +8.
__device__ __forceinline__ v16h frag_ld(const _Float16* base_, int row_base,
                                        int stride, int kbase, int lane) {
  const _Float16* p =
      base_ + (row_base + (lane & 15)) * stride + kbase + ((lane >> 4) << 3);
  v8h lo = *(const v8h*)(p);
  v8h hi = *(const v8h*)(p + 16);
  v16h f;
#pragma unroll
  for (int t = 0; t < 8; ++t) { f[t] = lo[t]; f[t + 8] = hi[t]; }
  return f;
}

// W fragment straight from global f32 (L2-resident): B[k][n] = W[colg+n][k],
// contiguous in k per lane, converted to f16 on the fly.
__device__ __forceinline__ v16h frag_ld_w(const float* __restrict__ w, int colg,
                                          int kbase, int lane) {
  const float* p =
      w + (size_t)(colg + (lane & 15)) * DIM + kbase + ((lane >> 4) << 3);
  v4f a0 = *(const v4f*)(p);
  v4f a1 = *(const v4f*)(p + 4);
  v4f b0 = *(const v4f*)(p + 16);
  v4f b1 = *(const v4f*)(p + 20);
  v16h f;
#pragma unroll
  for (int t = 0; t < 4; ++t) {
    f[t]      = (_Float16)a0[t];
    f[t + 4]  = (_Float16)a1[t];
    f[t + 8]  = (_Float16)b0[t];
    f[t + 12] = (_Float16)b1[t];
  }
  return f;
}

// XOR swap across each 16-lane half of the wave32 via ds_swizzle
// (group-of-32 mode: offset = xor_mask<<10 | and_mask 0x1f).
template <int MASK>
__device__ __forceinline__ float swz(float x) {
  return __int_as_float(
      __builtin_amdgcn_ds_swizzle(__float_as_int(x), (MASK << 10) | 0x1f));
}

__global__ __launch_bounds__(256, 1) void win_attn_proj_kernel(
    const float* __restrict__ q, const float* __restrict__ kk,
    const float* __restrict__ vv, const float* __restrict__ w,
    const float* __restrict__ pb, const float* __restrict__ bt,
    const int* __restrict__ ri, float* __restrict__ out) {
  __shared__ __align__(16) _Float16 smem[SMEM_HALVES];
  _Float16* sQ  = smem + SQ_OFF;
  _Float16* sK  = smem + SK_OFF;
  _Float16* sP  = smem + SP_OFF;   // valid only after the post-phase-1 barrier
  _Float16* sVt = smem + SVT_OFF;
  _Float16* sY  = smem + SY_OFF;

  const int tid  = threadIdx.x;
  const int lane = tid & 31;
  const int h    = tid >> 5;  // wave == head
  const size_t base = (size_t)blockIdx.x * (N_TOK * DIM);

  // ---------------- Phase 0: stage q,k row-major f16; v transposed --------
  // Streamed-once data: nontemporal loads keep L2 free for proj_w reuse.
  const v4f* q4 = (const v4f*)(q + base);
  const v4f* k4 = (const v4f*)(kk + base);
  const v4f* v4 = (const v4f*)(vv + base);
#pragma unroll
  for (int t = 0; t < 16; ++t) {
    const int it  = tid + t * 256;
    const int row = it >> 6;    // 64 v4f per 256-wide row
    const int c4  = it & 63;
    v4f a = __builtin_nontemporal_load((const v4f*)q4 + it);
    v4f b = __builtin_nontemporal_load((const v4f*)k4 + it);
    v4f c = __builtin_nontemporal_load((const v4f*)v4 + it);
    v4h ha = {(_Float16)a[0], (_Float16)a[1], (_Float16)a[2], (_Float16)a[3]};
    v4h hb = {(_Float16)b[0], (_Float16)b[1], (_Float16)b[2], (_Float16)b[3]};
    *(v4h*)&sQ[row * 264 + c4 * 4] = ha;
    *(v4h*)&sK[row * 264 + c4 * 4] = hb;
    const int col = c4 * 4;
    sVt[(col + 0) * 72 + row] = (_Float16)c[0];
    sVt[(col + 1) * 72 + row] = (_Float16)c[1];
    sVt[(col + 2) * 72 + row] = (_Float16)c[2];
    sVt[(col + 3) * 72 + row] = (_Float16)c[3];
  }
  __syncthreads();

  // ---------------- Phase 1: S = scale * Q K^T ; softmax ; + bias ---------
  // P kept in registers (f16 pairs packed in u32) until sQ/sK are dead,
  // then spilled to sP which overlays them.
  unsigned preg[4][16];  // [i][j*4 + rpair]
  const float scale = 0.17677669529663687f;  // 32^-0.5
  const int nl = lane & 15;
  const int hb = (lane >> 4) << 3;
#pragma unroll
  for (int i = 0; i < 4; ++i) {
    v16h afrag = frag_ld(sQ, 16 * i, 264, h * DH, lane);
    v8f c0 = {}, c1 = {}, c2 = {}, c3 = {};
    c0 = wmma16(afrag, frag_ld(sK, 0,  264, h * DH, lane), c0);
    c1 = wmma16(afrag, frag_ld(sK, 16, 264, h * DH, lane), c1);
    c2 = wmma16(afrag, frag_ld(sK, 32, 264, h * DH, lane), c2);
    c3 = wmma16(afrag, frag_ld(sK, 48, 264, h * DH, lane), c3);
    const int Mbase = 16 * i + hb;

    // Hoist all rel_index gathers for this tile; latency overlaps the
    // softmax reduction tree below.
    int rv0[8], rv1[8], rv2[8], rv3[8];
#pragma unroll
    for (int r = 0; r < 8; ++r) {
      const int rowoff = (Mbase + r) * 64 + nl;
      rv0[r] = ri[rowoff];
      rv1[r] = ri[rowoff + 16];
      rv2[r] = ri[rowoff + 32];
      rv3[r] = ri[rowoff + 48];
    }

    float s0[8], s1[8], s2[8], s3[8], mx[8], sum[8];
#pragma unroll
    for (int r = 0; r < 8; ++r) {
      s0[r] = c0[r] * scale;  s1[r] = c1[r] * scale;
      s2[r] = c2[r] * scale;  s3[r] = c3[r] * scale;
      mx[r] = fmaxf(fmaxf(s0[r], s1[r]), fmaxf(s2[r], s3[r]));
    }
    // Batched XOR-tree max: 8 independent swizzle chains per step.
#pragma unroll
    for (int r = 0; r < 8; ++r) mx[r] = fmaxf(mx[r], swz<1>(mx[r]));
#pragma unroll
    for (int r = 0; r < 8; ++r) mx[r] = fmaxf(mx[r], swz<2>(mx[r]));
#pragma unroll
    for (int r = 0; r < 8; ++r) mx[r] = fmaxf(mx[r], swz<4>(mx[r]));
#pragma unroll
    for (int r = 0; r < 8; ++r) mx[r] = fmaxf(mx[r], swz<8>(mx[r]));
#pragma unroll
    for (int r = 0; r < 8; ++r) {
      s0[r] = __expf(s0[r] - mx[r]);  s1[r] = __expf(s1[r] - mx[r]);
      s2[r] = __expf(s2[r] - mx[r]);  s3[r] = __expf(s3[r] - mx[r]);
      sum[r] = (s0[r] + s1[r]) + (s2[r] + s3[r]);
    }
#pragma unroll
    for (int r = 0; r < 8; ++r) sum[r] += swz<1>(sum[r]);
#pragma unroll
    for (int r = 0; r < 8; ++r) sum[r] += swz<2>(sum[r]);
#pragma unroll
    for (int r = 0; r < 8; ++r) sum[r] += swz<4>(sum[r]);
#pragma unroll
    for (int r = 0; r < 8; ++r) sum[r] += swz<8>(sum[r]);

    float inv[8];
#pragma unroll
    for (int r = 0; r < 8; ++r) inv[r] = __builtin_amdgcn_rcpf(sum[r]);

    // Bias gathered and added AFTER softmax (faithful to reference); pack
    // adjacent rows into one u32 (v_pack_b32_f16) to halve P's reg footprint.
#pragma unroll
    for (int rp = 0; rp < 4; ++rp) {
      const int re = 2 * rp, ro = 2 * rp + 1;
      const float b0e = bt[rv0[re] * NH + h], b0o = bt[rv0[ro] * NH + h];
      const float b1e = bt[rv1[re] * NH + h], b1o = bt[rv1[ro] * NH + h];
      const float b2e = bt[rv2[re] * NH + h], b2o = bt[rv2[ro] * NH + h];
      const float b3e = bt[rv3[re] * NH + h], b3o = bt[rv3[ro] * NH + h];
      union { _Float16 hh[2]; unsigned u; } p0, p1, p2, p3;
      p0.hh[0] = (_Float16)(s0[re] * inv[re] + b0e);
      p0.hh[1] = (_Float16)(s0[ro] * inv[ro] + b0o);
      p1.hh[0] = (_Float16)(s1[re] * inv[re] + b1e);
      p1.hh[1] = (_Float16)(s1[ro] * inv[ro] + b1o);
      p2.hh[0] = (_Float16)(s2[re] * inv[re] + b2e);
      p2.hh[1] = (_Float16)(s2[ro] * inv[ro] + b2o);
      p3.hh[0] = (_Float16)(s3[re] * inv[re] + b3e);
      p3.hh[1] = (_Float16)(s3[ro] * inv[ro] + b3o);
      preg[i][0 * 4 + rp] = p0.u;
      preg[i][1 * 4 + rp] = p1.u;
      preg[i][2 * 4 + rp] = p2.u;
      preg[i][3 * 4 + rp] = p3.u;
    }
  }
  __syncthreads();  // everyone done reading sQ/sK -> safe to overlay with sP
#pragma unroll
  for (int i = 0; i < 4; ++i)
#pragma unroll
    for (int j = 0; j < 4; ++j)
#pragma unroll
      for (int rp = 0; rp < 4; ++rp) {
        union { unsigned u; _Float16 hh[2]; } pu;
        pu.u = preg[i][j * 4 + rp];
        const int M = 16 * i + hb + 2 * rp;
        sP[h * 4608 + (M + 0) * 72 + 16 * j + nl] = pu.hh[0];
        sP[h * 4608 + (M + 1) * 72 + 16 * j + nl] = pu.hh[1];
      }
  __syncthreads();

  // ---------------- Phase 2: O = P @ V  (per head) ------------------------
#pragma unroll
  for (int jt = 0; jt < 2; ++jt) {
    v8f o0 = {}, o1 = {}, o2 = {}, o3 = {};
#pragma unroll
    for (int kc = 0; kc < 2; ++kc) {
      v16h bfrag = frag_ld(sVt, h * DH + jt * 16, 72, kc * 32, lane);
      o0 = wmma16(frag_ld(sP + h * 4608, 0,  72, kc * 32, lane), bfrag, o0);
      o1 = wmma16(frag_ld(sP + h * 4608, 16, 72, kc * 32, lane), bfrag, o1);
      o2 = wmma16(frag_ld(sP + h * 4608, 32, 72, kc * 32, lane), bfrag, o2);
      o3 = wmma16(frag_ld(sP + h * 4608, 48, 72, kc * 32, lane), bfrag, o3);
    }
    const int colb = h * DH + jt * 16 + nl;
#pragma unroll
    for (int r = 0; r < 8; ++r) {
      sY[(0  + hb + r) * 264 + colb] = (_Float16)o0[r];
      sY[(16 + hb + r) * 264 + colb] = (_Float16)o1[r];
      sY[(32 + hb + r) * 264 + colb] = (_Float16)o2[r];
      sY[(48 + hb + r) * 264 + colb] = (_Float16)o3[r];
    }
  }
  __syncthreads();

  // ---------------- Phase 3: Z = Y @ W^T + b ------------------------------
  float* outp = out + base;
#pragma unroll
  for (int cti = 0; cti < 2; ++cti) {
    const int colg = (h * 2 + cti) * 16;
    v8f z0 = {}, z1 = {}, z2 = {}, z3 = {};
#pragma unroll
    for (int kc = 0; kc < 8; ++kc) {
      v16h bfrag = frag_ld_w(w, colg, kc * 32, lane);
      z0 = wmma16(frag_ld(sY, 0,  264, kc * 32, lane), bfrag, z0);
      z1 = wmma16(frag_ld(sY, 16, 264, kc * 32, lane), bfrag, z1);
      z2 = wmma16(frag_ld(sY, 32, 264, kc * 32, lane), bfrag, z2);
      z3 = wmma16(frag_ld(sY, 48, 264, kc * 32, lane), bfrag, z3);
    }
    const float bc = pb[colg + nl];
#pragma unroll
    for (int r = 0; r < 8; ++r) {
      __builtin_nontemporal_store(z0[r] + bc,
          outp + (size_t)(0  + hb + r) * DIM + colg + nl);
      __builtin_nontemporal_store(z1[r] + bc,
          outp + (size_t)(16 + hb + r) * DIM + colg + nl);
      __builtin_nontemporal_store(z2[r] + bc,
          outp + (size_t)(32 + hb + r) * DIM + colg + nl);
      __builtin_nontemporal_store(z3[r] + bc,
          outp + (size_t)(48 + hb + r) * DIM + colg + nl);
    }
  }
}

extern "C" void kernel_launch(void* const* d_in, const int* in_sizes, int n_in,
                              void* d_out, int out_size, void* d_ws,
                              size_t ws_size, hipStream_t stream) {
  (void)n_in; (void)out_size; (void)d_ws; (void)ws_size;
  // setup_inputs order: x, q, k, v, proj_w, proj_b, bias_table, rel_index
  const float* q  = (const float*)d_in[1];
  const float* k  = (const float*)d_in[2];
  const float* v  = (const float*)d_in[3];
  const float* w  = (const float*)d_in[4];
  const float* pb = (const float*)d_in[5];
  const float* bt = (const float*)d_in[6];
  const int*   ri = (const int*)d_in[7];
  float* out = (float*)d_out;
  const int B = in_sizes[1] / (N_TOK * DIM);  // 2048
  win_attn_proj_kernel<<<dim3(B), dim3(256), 0, stream>>>(q, k, v, w, pb, bt,
                                                          ri, out);
}
